// PixelAPLoss_61409442398790
// MI455X (gfx1250) — matrix operand order repaired and checked by reading, not compiled, not run on hardware
//
#include <hip/hip_runtime.h>
#include <hip/hip_bf16.h>

#define NQ_   20
#define NEG_  80
#define NSC_  81
#define PMAX_ (480 * 640 - 1)
#define DDIM  64

typedef __attribute__((ext_vector_type(16))) _Float16 v16h;
typedef __attribute__((ext_vector_type(8)))  float    v8f;

// One wave per match point n. Computes the 81 scores with WMMA (6 tiles of 16
// rows, K=64 as two chained 16x16x32 f16 WMMAs with B = a1 broadcast across
// all 16 columns), then the quantized-AP epilogue, writing (1 - ap_n) to ws.
__launch_bounds__(32)
__global__ void pixel_ap_kernel(const float* __restrict__ d1,
                                const float* __restrict__ d2,
                                const float* __restrict__ qw,
                                const float* __restrict__ qb,
                                const int*   __restrict__ m1,
                                const int*   __restrict__ m2,
                                const int*   __restrict__ offs,
                                float*       __restrict__ per_n)
{
    __shared__ _Float16 aTile[16 * DDIM];  // 16 gathered d2 rows, f16
    __shared__ _Float16 bvec[DDIM];        // a1 in f16
    __shared__ float    scores[96];
    __shared__ int      offs_s[NEG_];
    __shared__ float    cr[NQ_], cn[NQ_];

    const int n     = blockIdx.x;
    const int lane  = threadIdx.x;
    const int base2 = m2[n];

    // Stage offsets once.
    for (int i = lane; i < NEG_; i += 32) offs_s[i] = offs[i];

    // Gather a1 (64 f32 -> f16), 2 elements per lane (coalesced b64 loads).
    {
        const float2 v = *(const float2*)(d1 + (size_t)m1[n] * DDIM + lane * 2);
        bvec[lane * 2 + 0] = (_Float16)v.x;
        bvec[lane * 2 + 1] = (_Float16)v.y;
    }
    __syncthreads();

    // B fragments: B[k][j] = a1[k] for all 16 columns j (broadcast), so C's
    // column 0 carries the scores. B k order used here (contiguous, same as A): lanes 0-15 hold
    // K=0..15, lanes 16-31 hold K=16..31; VGPR v holds K=kb+2v (lo), kb+2v+1 (hi).
    v16h bf0 = {}, bf1 = {};
    {
        const int kb = (lane < 16) ? 0 : 16;
#pragma unroll
        for (int v = 0; v < 8; ++v) {
            bf0[2 * v + 0] = bvec[kb + 2 * v + 0];
            bf0[2 * v + 1] = bvec[kb + 2 * v + 1];
            bf1[2 * v + 0] = bvec[32 + kb + 2 * v + 0];
            bf1[2 * v + 1] = bvec[32 + kb + 2 * v + 1];
        }
    }

    for (int t = 0; t < 6; ++t) {
        // Gather 16 d2 rows (f32 -> f16). Each lane pair splits one row:
        // lane handles row (lane>>1), 32 columns at (lane&1)*32, via 8x b128.
        {
            const int i   = lane >> 1;
            const int r   = t * 16 + i;                       // global score row
            const int off = (r >= 1 && r <= NEG_) ? offs_s[r - 1] : 0;
            int g = base2 + off;                              // r==0 -> positive row
            g = (g < 0) ? 0 : ((g > PMAX_) ? PMAX_ : g);
            const float4* rp  = (const float4*)(d2 + (size_t)g * DDIM + (lane & 1) * 32);
            _Float16*     dst = &aTile[i * DDIM + (lane & 1) * 32];
#pragma unroll
            for (int q4 = 0; q4 < 8; ++q4) {
                const float4 v = rp[q4];
                dst[4 * q4 + 0] = (_Float16)v.x;
                dst[4 * q4 + 1] = (_Float16)v.y;
                dst[4 * q4 + 2] = (_Float16)v.z;
                dst[4 * q4 + 3] = (_Float16)v.w;
            }
        }
        __syncthreads();

        // A fragments in contiguous k order (k = 16*(lane/16) + element), matching B:
        // lanes 0-15: M=lane, VGPR0-3 -> K=0..7,  VGPR4-7 -> K=8..15
        // lanes 16-31: M=lane-16, VGPR0-3 -> K=16..23, VGPR4-7 -> K=24..31
        const int Mrow = lane & 15;
        const int kb   = (lane < 16) ? 0 : 16;
        v16h af0 = {}, af1 = {};
#pragma unroll
        for (int v = 0; v < 4; ++v) {
#pragma unroll
            for (int h = 0; h < 2; ++h) {
                af0[2 * v + h]       = aTile[Mrow * DDIM + (kb + 2 * v + h)];
                af0[2 * (v + 4) + h] = aTile[Mrow * DDIM + (8 + kb + 2 * v + h)];
                af1[2 * v + h]       = aTile[Mrow * DDIM + (32 + kb + 2 * v + h)];
                af1[2 * (v + 4) + h] = aTile[Mrow * DDIM + (40 + kb + 2 * v + h)];
            }
        }

        v8f c = {};
        c = __builtin_amdgcn_wmma_f32_16x16x32_f16(false, af0, false, bf0,
                                                   (short)0, c, false, false);
        c = __builtin_amdgcn_wmma_f32_16x16x32_f16(false, af1, false, bf1,
                                                   (short)0, c, false, false);

        // Column 0 of C: lane 0 holds rows 0..7 in c[0..7], lane 16 rows 8..15.
        if (lane == 0) {
#pragma unroll
            for (int i = 0; i < 8; ++i) scores[t * 16 + i] = c[i];
        } else if (lane == 16) {
#pragma unroll
            for (int i = 0; i < 8; ++i) scores[t * 16 + 8 + i] = c[i];
        }
        __syncthreads();
    }

    // Quantizer: lanes 0..19 each own one bin over the 81 scores.
    if (lane < NQ_) {
        const float w1 = qw[lane],        b1 = qb[lane];
        const float w2 = qw[lane + NQ_],  b2 = qb[lane + NQ_];
        float nbs = 0.f, rec0 = 0.f;
#pragma unroll 1
        for (int mi = 0; mi < NSC_; ++mi) {
            const float x = scores[mi];
            float q = fminf(fmaf(w1, x, b1), fmaf(w2, x, b2));
            q = fmaxf(q, 0.f);
            nbs += q;
            if (mi == 0) rec0 = q;
        }
        cr[lane] = rec0;   // rec[k]  = q[k][0]
        cn[lane] = nbs;    // nbs[k]  = sum_m q[k][m]
    }
    __syncthreads();

    // AP over the 20 bins (serial on lane 0; 20 elements, negligible).
    if (lane == 0) {
        float crun = 0.f, nrun = 0.f, rsum = 0.f, apn = 0.f;
        float prec[NQ_];
#pragma unroll
        for (int k = 0; k < NQ_; ++k) {
            crun += cr[k];
            nrun += cn[k];
            prec[k] = crun / (1e-16f + nrun);
            rsum += cr[k];
        }
#pragma unroll
        for (int k = 0; k < NQ_; ++k) apn += prec[k] * (cr[k] / rsum);
        per_n[n] = 1.f - apn;
    }
}

// Deterministic fixed-order mean reduction (no float atomics).
__launch_bounds__(256)
__global__ void reduce_mean_kernel(const float* __restrict__ per_n,
                                   float* __restrict__ out, int N)
{
    __shared__ float s[256];
    const int t = threadIdx.x;
    float acc = 0.f;
    for (int i = t; i < N; i += 256) acc += per_n[i];
    s[t] = acc;
    __syncthreads();
    for (int w = 128; w > 0; w >>= 1) {
        if (t < w) s[t] += s[t + w];
        __syncthreads();
    }
    if (t == 0) out[0] = s[0] / (float)N;
}

extern "C" void kernel_launch(void* const* d_in, const int* in_sizes, int n_in,
                              void* d_out, int out_size, void* d_ws, size_t ws_size,
                              hipStream_t stream)
{
    const float* d1   = (const float*)d_in[0];  // descriptors1 (1,P,64)
    const float* d2   = (const float*)d_in[1];  // descriptors2 (1,P,64)
    const float* qw   = (const float*)d_in[2];  // (40,)
    const float* qb   = (const float*)d_in[3];  // (40,)
    const int*   m1   = (const int*)  d_in[4];  // (N,)
    const int*   m2   = (const int*)  d_in[5];  // (N,)
    const int*   offs = (const int*)  d_in[6];  // (80,)
    const int N = in_sizes[4];

    float* per_n = (float*)d_ws;  // N floats of scratch

    pixel_ap_kernel<<<N, 32, 0, stream>>>(d1, d2, qw, qb, m1, m2, offs, per_n);
    reduce_mean_kernel<<<1, 256, 0, stream>>>(per_n, (float*)d_out, N);
}
